// GRUModel_17403207483537
// MI455X (gfx1250) — matrix operand order repaired and checked by reading, not compiled, not run on hardware
//
#include <hip/hip_runtime.h>
#include <math.h>

// Fully-fused 2-layer GRU + FC head for MI455X (gfx1250, wave32, WMMA).
//
// B=4096, T=256, IN=21, H=64, G=3H=192.
// Each block: 16 batch rows through all 256 steps of BOTH layers (h1,h2 stay
// on-chip -> no 268MB intermediate; HBM traffic ~= x (88MB) + out).
// 128 threads = 4 wave32; wave w owns gate-column tile j=w (cols j*16 of each
// of r/z/n). Matrix core: v_wmma_f32_16x16x32_bf16. Recurrent weights (Whh0,
// Whh1) are pre-fragmented into VGPRs once; Wih0/Wih1 fragments stream from LDS.
// Activations use hardware transcendentals (v_exp_f32 / v_tanh_f32 / v_rcp_f32)
// so the step body is branch-free (EXEC stays all-ones around every WMMA).

#define B_TOT    4096
#define T_LEN    256
#define IN_SZ    21
#define HID      64
#define ROWS     16      // batch rows per block
#define NTHREADS 128     // 4 waves

typedef __attribute__((ext_vector_type(16))) __bf16 bf16x16;
typedef __attribute__((ext_vector_type(8)))  __bf16 bf16x8;
typedef __attribute__((ext_vector_type(8)))  float  f32x8;

// A-fragment (16x32 bf16, M rows = tile rows 0..15 of a row-major LDS matrix).
// Layout per ISA: lanes 0-15 hold K 0..7 (v0-3) and 16..23 (v4-7);
// lanes 16-31 hold K 8..15 and 24..31. Both halves are contiguous 16B runs.
__device__ __forceinline__ bf16x16 ldsA(const __bf16* base, int ld, int k0, int lane) {
  const int half = lane >> 4, mr = lane & 15;
  const __bf16* p = base + mr * ld + k0 + half * 8;
  bf16x8 lo = *(const bf16x8*)(p);
  bf16x8 hi = *(const bf16x8*)(p + 16);
  return __builtin_shufflevector(lo, hi, 0,1,2,3,4,5,6,7,8,9,10,11,12,13,14,15);
}

// B-fragment (32x16 bf16). We compute C[m,g] = sum_k A[m,k]*W[g,k], so the
// B operand's column n maps to row g=n0+lane&15 of the row-major weight matrix;
// lanes 0-15 take K k0..k0+7 and k0+16..k0+23, lanes 16-31 take k0+8..k0+15 and k0+24..k0+31.
__device__ __forceinline__ bf16x16 ldsB(const __bf16* base, int ld, int n0, int k0, int lane) {
  const int half = lane >> 4, c = lane & 15;
  const __bf16* p = base + (n0 + c) * ld + k0 + half * 8;
  bf16x8 lo = *(const bf16x8*)(p);
  bf16x8 hi = *(const bf16x8*)(p + 16);
  return __builtin_shufflevector(lo, hi, 0,1,2,3,4,5,6,7,8,9,10,11,12,13,14,15);
}

__device__ __forceinline__ f32x8 wmma_bf16(bf16x16 a, bf16x16 b, f32x8 c) {
  return __builtin_amdgcn_wmma_f32_16x16x32_bf16(
      /*neg_a=*/false, a, /*neg_b=*/false, b,
      /*c_mod=*/(short)0, c, /*reuse_a=*/false, /*reuse_b=*/false);
}

// sigmoid(x) = rcp(1 + exp(-x)) : v_mul + v_exp_f32 + v_add + v_rcp_f32
__device__ __forceinline__ float fast_sigmoid(float x) {
  return __builtin_amdgcn_rcpf(1.0f + __expf(-x));
}

// tanh: gfx1250 has V_TANH_F32; fall back to a branch-free exp2/rcp form.
__device__ __forceinline__ float fast_tanh(float x) {
#if __has_builtin(__builtin_amdgcn_tanhf)
  return __builtin_amdgcn_tanhf(x);
#else
  // tanh(x) = 1 - 2/(exp(2x)+1)
  float e = __expf(2.0f * x);
  return 1.0f - 2.0f * __builtin_amdgcn_rcpf(e + 1.0f);
#endif
}

__device__ __forceinline__ f32x8 splat8(float v) {
  f32x8 r;
#pragma unroll
  for (int i = 0; i < 8; ++i) r[i] = v;
  return r;
}

__global__ __launch_bounds__(NTHREADS)
void gru2_fused_kernel(const float* __restrict__ x,
                       const float* __restrict__ Wih0, const float* __restrict__ Whh0,
                       const float* __restrict__ bih0, const float* __restrict__ bhh0,
                       const float* __restrict__ Wih1, const float* __restrict__ Whh1,
                       const float* __restrict__ bih1, const float* __restrict__ bhh1,
                       const float* __restrict__ Wfc,  const float* __restrict__ bfc,
                       float* __restrict__ out)
{
  extern __shared__ char smem[];
  __bf16* Wih0L = (__bf16*)smem;            // [192][32]  (K padded 21 -> 32 with zeros)
  __bf16* Whh0L = Wih0L + 192 * 32;         // [192][64]
  __bf16* Wih1L = Whh0L + 192 * 64;         // [192][64]
  __bf16* Whh1L = Wih1L + 192 * 64;         // [192][64]
  __bf16* xbuf  = Whh1L + 192 * 64;         // [16][32]   (cols >= 21 stay zero)
  __bf16* h1L   = xbuf  + ROWS * 32;        // [16][64]
  __bf16* h2L   = h1L   + ROWS * 64;        // [16][64]

  const int tid  = threadIdx.x;
  const int lane = tid & 31;
  const int j    = tid >> 5;        // wave id == gate-column tile (0..3)
  const int c    = lane & 15;       // this lane's column within the tile
  const int half = lane >> 4;
  const int b0   = blockIdx.x * ROWS;

  // ---- stage weights to LDS (f32 -> bf16), zero pads & states ----
  for (int idx = tid; idx < 192 * 32; idx += NTHREADS) {
    int r = idx >> 5, cc = idx & 31;
    Wih0L[idx] = (cc < IN_SZ) ? (__bf16)Wih0[r * IN_SZ + cc] : (__bf16)0.0f;
  }
  for (int idx = tid; idx < 192 * 64; idx += NTHREADS) Whh0L[idx] = (__bf16)Whh0[idx];
  for (int idx = tid; idx < 192 * 64; idx += NTHREADS) Wih1L[idx] = (__bf16)Wih1[idx];
  for (int idx = tid; idx < 192 * 64; idx += NTHREADS) Whh1L[idx] = (__bf16)Whh1[idx];
  for (int idx = tid; idx < ROWS * 32; idx += NTHREADS) xbuf[idx] = (__bf16)0.0f;
  for (int idx = tid; idx < ROWS * 64; idx += NTHREADS) {
    h1L[idx] = (__bf16)0.0f;
    h2L[idx] = (__bf16)0.0f;
  }
  __syncthreads();

  // ---- per-lane gate biases (constant over t) ----
  const float br1  = bih0[      j*16 + c] + bhh0[      j*16 + c];
  const float bz1  = bih0[ 64 + j*16 + c] + bhh0[ 64 + j*16 + c];
  const float bxn1 = bih0[128 + j*16 + c];
  const float bhn1 = bhh0[128 + j*16 + c];
  const float br2  = bih1[      j*16 + c] + bhh1[      j*16 + c];
  const float bz2  = bih1[ 64 + j*16 + c] + bhh1[ 64 + j*16 + c];
  const float bxn2 = bih1[128 + j*16 + c];
  const float bhn2 = bhh1[128 + j*16 + c];

  // ---- pre-fragment the recurrent weights into VGPRs (12 frags, constant) ----
  const bf16x16 W0r0 = ldsB(Whh0L, 64,       j*16,  0, lane);
  const bf16x16 W0r1 = ldsB(Whh0L, 64,       j*16, 32, lane);
  const bf16x16 W0z0 = ldsB(Whh0L, 64,  64 + j*16,  0, lane);
  const bf16x16 W0z1 = ldsB(Whh0L, 64,  64 + j*16, 32, lane);
  const bf16x16 W0n0 = ldsB(Whh0L, 64, 128 + j*16,  0, lane);
  const bf16x16 W0n1 = ldsB(Whh0L, 64, 128 + j*16, 32, lane);
  const bf16x16 W1r0 = ldsB(Whh1L, 64,       j*16,  0, lane);
  const bf16x16 W1r1 = ldsB(Whh1L, 64,       j*16, 32, lane);
  const bf16x16 W1z0 = ldsB(Whh1L, 64,  64 + j*16,  0, lane);
  const bf16x16 W1z1 = ldsB(Whh1L, 64,  64 + j*16, 32, lane);
  const bf16x16 W1n0 = ldsB(Whh1L, 64, 128 + j*16,  0, lane);
  const bf16x16 W1n1 = ldsB(Whh1L, 64, 128 + j*16, 32, lane);

  // h_old in f32 registers, C-tile layout (this wave reproduces the same tile
  // every step, so the producer is the consumer; LDS copy feeds WMMA A-frags).
  float h1reg[8], h2reg[8];
#pragma unroll
  for (int i = 0; i < 8; ++i) { h1reg[i] = 0.0f; h2reg[i] = 0.0f; }

  const float* xblk = x + (size_t)b0 * T_LEN * IN_SZ;

#pragma unroll 1
  for (int t = 0; t < T_LEN; ++t) {
    // ---- stage x_t tile (16 x 21 f32 -> bf16, pad cols stay zero) ----
    for (int idx = tid; idx < ROWS * IN_SZ; idx += NTHREADS) {
      int r = idx / IN_SZ, cc = idx - r * IN_SZ;
      xbuf[r * 32 + cc] =
          (__bf16)xblk[(size_t)r * T_LEN * IN_SZ + (size_t)t * IN_SZ + cc];
    }
    // hint the next step's input toward L2 early
    if (t + 1 < T_LEN && tid < ROWS)
      __builtin_prefetch(&xblk[(size_t)tid * T_LEN * IN_SZ + (size_t)(t + 1) * IN_SZ], 0, 1);
    __syncthreads();

    // =================== layer 1 ===================
    bf16x16 Ax  = ldsA(xbuf, 32,  0, lane);
    bf16x16 Ah0 = ldsA(h1L,  64,  0, lane);
    bf16x16 Ah1 = ldsA(h1L,  64, 32, lane);
    f32x8 ar  = splat8(br1);
    f32x8 az  = splat8(bz1);
    f32x8 axn = splat8(bxn1);
    f32x8 ahn = splat8(bhn1);
    ar  = wmma_bf16(Ax,  ldsB(Wih0L, 32,       j*16, 0, lane), ar);
    ar  = wmma_bf16(Ah0, W0r0, ar);
    ar  = wmma_bf16(Ah1, W0r1, ar);
    az  = wmma_bf16(Ax,  ldsB(Wih0L, 32,  64 + j*16, 0, lane), az);
    az  = wmma_bf16(Ah0, W0z0, az);
    az  = wmma_bf16(Ah1, W0z1, az);
    axn = wmma_bf16(Ax,  ldsB(Wih0L, 32, 128 + j*16, 0, lane), axn);
    ahn = wmma_bf16(Ah0, W0n0, ahn);
    ahn = wmma_bf16(Ah1, W0n1, ahn);
    __syncthreads();   // all waves finished reading h1L (old)

#pragma unroll
    for (int i = 0; i < 8; ++i) {
      float rg = fast_sigmoid(ar[i]);
      float zg = fast_sigmoid(az[i]);
      float ng = fast_tanh(axn[i] + rg * ahn[i]);
      float h  = (1.0f - zg) * ng + zg * h1reg[i];
      h1reg[i] = h;
      h1L[(i + half * 8) * 64 + j * 16 + c] = (__bf16)h;   // C-layout scatter
    }
    __syncthreads();   // h1 (new) visible

    // =================== layer 2 ===================
    bf16x16 Bx0 = ldsA(h1L, 64,  0, lane);   // layer-2 "x" = h1_new
    bf16x16 Bx1 = ldsA(h1L, 64, 32, lane);
    bf16x16 Bh0 = ldsA(h2L, 64,  0, lane);
    bf16x16 Bh1 = ldsA(h2L, 64, 32, lane);
    f32x8 ar2  = splat8(br2);
    f32x8 az2  = splat8(bz2);
    f32x8 axn2 = splat8(bxn2);
    f32x8 ahn2 = splat8(bhn2);
    ar2  = wmma_bf16(Bx0, ldsB(Wih1L, 64,       j*16,  0, lane), ar2);
    ar2  = wmma_bf16(Bx1, ldsB(Wih1L, 64,       j*16, 32, lane), ar2);
    ar2  = wmma_bf16(Bh0, W1r0, ar2);
    ar2  = wmma_bf16(Bh1, W1r1, ar2);
    az2  = wmma_bf16(Bx0, ldsB(Wih1L, 64,  64 + j*16,  0, lane), az2);
    az2  = wmma_bf16(Bx1, ldsB(Wih1L, 64,  64 + j*16, 32, lane), az2);
    az2  = wmma_bf16(Bh0, W1z0, az2);
    az2  = wmma_bf16(Bh1, W1z1, az2);
    axn2 = wmma_bf16(Bx0, ldsB(Wih1L, 64, 128 + j*16,  0, lane), axn2);
    axn2 = wmma_bf16(Bx1, ldsB(Wih1L, 64, 128 + j*16, 32, lane), axn2);
    ahn2 = wmma_bf16(Bh0, W1n0, ahn2);
    ahn2 = wmma_bf16(Bh1, W1n1, ahn2);
    __syncthreads();   // all waves finished reading h2L (old)

#pragma unroll
    for (int i = 0; i < 8; ++i) {
      float rg = fast_sigmoid(ar2[i]);
      float zg = fast_sigmoid(az2[i]);
      float ng = fast_tanh(axn2[i] + rg * ahn2[i]);
      float h  = (1.0f - zg) * ng + zg * h2reg[i];
      h2reg[i] = h;
      h2L[(i + half * 8) * 64 + j * 16 + c] = (__bf16)h;
    }
    __syncthreads();   // h2 (new) visible before next step / FC
  }

  // ---- FC head: out[b] = h2_last[b,:] . Wfc + bfc ----
  if (tid < ROWS) {
    float s = bfc[0];
#pragma unroll
    for (int k = 0; k < HID; ++k) s += (float)h2L[tid * 64 + k] * Wfc[k];
    out[b0 + tid] = s;
  }
}

extern "C" void kernel_launch(void* const* d_in, const int* in_sizes, int n_in,
                              void* d_out, int out_size, void* d_ws, size_t ws_size,
                              hipStream_t stream) {
  (void)in_sizes; (void)n_in; (void)out_size; (void)d_ws; (void)ws_size;
  const float* x    = (const float*)d_in[0];
  const float* Wih0 = (const float*)d_in[1];
  const float* Whh0 = (const float*)d_in[2];
  const float* bih0 = (const float*)d_in[3];
  const float* bhh0 = (const float*)d_in[4];
  const float* Wih1 = (const float*)d_in[5];
  const float* Whh1 = (const float*)d_in[6];
  const float* bih1 = (const float*)d_in[7];
  const float* bhh1 = (const float*)d_in[8];
  const float* Wfc  = (const float*)d_in[9];
  const float* bfc  = (const float*)d_in[10];
  float* out = (float*)d_out;

  // dynamic LDS: weights (bf16) + x tile + h1 + h2
  const size_t ldsBytes =
      (size_t)(192 * 32 + 3 * 192 * 64 + ROWS * 32 + 2 * ROWS * 64) * 2;  // 91,136 B

  (void)hipFuncSetAttribute((const void*)gru2_fused_kernel,
                            hipFuncAttributeMaxDynamicSharedMemorySize,
                            (int)ldsBytes);

  gru2_fused_kernel<<<B_TOT / ROWS, NTHREADS, ldsBytes, stream>>>(
      x, Wih0, Whh0, bih0, bhh0, Wih1, Whh1, bih1, bhh1, Wfc, bfc, out);
}
